// SaliencyTraceModule_13546326851793
// MI455X (gfx1250) — matrix operand order repaired and checked by reading, not compiled, not run on hardware
//
#include <hip/hip_runtime.h>
#include <cstdint>

// ---------------- problem constants (match reference) ----------------
#define B_   16
#define F_   2048
#define S_   512
#define D_   256
#define DLAT 512
#define D3   768   // 3*D
#define D2   512   // 2*D
#define EPB  32    // edges per block (2 M-tiles)

typedef __attribute__((ext_vector_type(16))) __bf16 v16bf;
typedef __attribute__((ext_vector_type(8)))  float  v8f;

union V16BF { uint32_t u[8]; uint4 q[2]; v16bf v; };

__device__ __forceinline__ unsigned short f2bf(float f) {
    unsigned int u = __float_as_uint(f);
    u += 0x7FFFu + ((u >> 16) & 1u);       // round-to-nearest-even
    return (unsigned short)(u >> 16);
}

__device__ __forceinline__ uint32_t pack2bf(float a, float b) {
    return (uint32_t)f2bf(a) | ((uint32_t)f2bf(b) << 16);
}

__device__ __forceinline__ float gelu_exact(float x) {
    return 0.5f * x * (1.0f + erff(x * 0.70710678118654752f));
}

__device__ __forceinline__ v8f vzero8() {
    v8f z = {0.f, 0.f, 0.f, 0.f, 0.f, 0.f, 0.f, 0.f};
    return z;
}

// ---------------------------------------------------------------------
// WMMA GEMM tile engine (rolled K-loop, software-pipelined B).
//   A: MT x (16 x KDIM) bf16 row-major in LDS (row stride KDIM/2 dwords)
//   W: [N][KDIM] bf16 row-major in global (pre-transposed)
//   acc[mt*NT+j] += A_tile(mt) @ W_tile(n0 + 16*j)
// Loop body = two K-steps with fixed buffer names (SROA-friendly);
// next iteration's B0 is prefetched inside the body; last iteration is
// peeled so there is no out-of-bounds prefetch. Addresses are a rolling
// base pointer + compile-time immediate offsets.
// ---------------------------------------------------------------------
template<int KDIM, int NT, int MT>
__device__ __forceinline__ void wmma_gemm(const uint32_t* __restrict__ Ad,
                                          const uint32_t* __restrict__ Wd,
                                          int n0, int lane, v8f* acc) {
    const int lhi = lane >> 4;          // 0: K-low half, 1: K-high half
    const int lm  = lane & 15;          // A row (M) / B column (N) within tile
    constexpr int AS = KDIM >> 1;       // row stride in dwords
    const uint4* wb = (const uint4*)(Wd + (size_t)(n0 + lm) * AS + (lhi ? 8 : 0));
    const uint4* ab = (const uint4*)(Ad + (size_t)lm * AS + (lhi ? 8 : 0));

    V16BF B0[NT], B1[NT];

    // o: uint4 offset of the K-step inside the current iteration window
#define WG_LOADB(BUF, o)                                                     \
    _Pragma("unroll")                                                        \
    for (int j = 0; j < NT; ++j) {                                           \
        BUF[j].q[0] = wb[j * 4 * AS + (o)];                                  \
        BUF[j].q[1] = wb[j * 4 * AS + (o) + 1];                              \
    }
#define WG_LOADA(AT, o)                                                      \
    _Pragma("unroll")                                                        \
    for (int mt = 0; mt < MT; ++mt) {                                        \
        AT[mt].q[0] = ab[mt * 4 * AS + (o)];                                 \
        AT[mt].q[1] = ab[mt * 4 * AS + (o) + 1];                             \
    }
#define WG_COMP(BUF, AT)                                                     \
    _Pragma("unroll")                                                        \
    for (int mt = 0; mt < MT; ++mt)                                          \
        _Pragma("unroll")                                                    \
        for (int j = 0; j < NT; ++j)                                         \
            acc[mt * NT + j] = __builtin_amdgcn_wmma_f32_16x16x32_bf16(      \
                false, AT[mt].v, false, BUF[j].v, (short)0,                  \
                acc[mt * NT + j], false, false);

    constexpr int ITERS = KDIM / 64;    // 64 K per iteration (two steps)
    WG_LOADB(B0, 0);
#pragma unroll 1
    for (int it = 0; it < ITERS - 1; ++it) {
        V16BF A0[MT], A1[MT];
        WG_LOADB(B1, 4);                // K-step +32 of this iteration
        WG_LOADA(A0, 0);
        WG_COMP(B0, A0);
        WG_LOADB(B0, 8);                // prefetch next iteration's first step
        WG_LOADA(A1, 4);
        WG_COMP(B1, A1);
        wb += 8;                        // advance 64 K (32 dwords = 8 uint4)
        ab += 8;
    }
    {   // peeled last iteration (no prefetch)
        V16BF A0[MT], A1[MT];
        WG_LOADB(B1, 4);
        WG_LOADA(A0, 0);
        WG_COMP(B0, A0);
        WG_LOADA(A1, 4);
        WG_COMP(B1, A1);
    }
#undef WG_LOADB
#undef WG_LOADA
#undef WG_COMP
}

// ---------------- prep kernels ----------------
__global__ __launch_bounds__(256) void zero_f32_kernel(float* p, int n) {
    int i = blockIdx.x * 256 + threadIdx.x;
    if (i < n) p[i] = 0.0f;
}

// dst[n][k] = bf16(src[k][n]); K, N multiples of 32. LDS-tiled transpose.
__global__ __launch_bounds__(256) void transpose_bf16_kernel(
        const float* __restrict__ src, unsigned short* __restrict__ dst, int K, int N) {
    __shared__ unsigned short tile[32][33];
    const int ntn = N >> 5;
    const int tk = blockIdx.x / ntn;
    const int tn = blockIdx.x - tk * ntn;
    const int k0 = tk << 5, n0 = tn << 5;
    const int c = threadIdx.x & 31;
    const int r0 = threadIdx.x >> 5;          // 8 rows per pass
#pragma unroll
    for (int r = 0; r < 32; r += 8)
        tile[r + r0][c] = f2bf(src[(size_t)(k0 + r + r0) * N + n0 + c]);
    __syncthreads();
#pragma unroll
    for (int r = 0; r < 32; r += 8)
        dst[(size_t)(n0 + r + r0) * K + k0 + c] = tile[c][r + r0];
}

// ---------------- edge-MLP kernel (32 edges / block) ----------------
__global__ __launch_bounds__(256) void edge_mlp_kernel(
        const int* __restrict__ a0, const int* __restrict__ a1,
        const int* __restrict__ pidx, const int* __restrict__ ridx,
        const float* __restrict__ pos, const float* __restrict__ pemb,
        const float* __restrict__ remb,
        const unsigned short* __restrict__ W1t, const float* __restrict__ b1,
        const unsigned short* __restrict__ W2t, const float* __restrict__ b2,
        float* __restrict__ agg) {
    __shared__ __align__(16) uint32_t Xd[EPB * (D3 / 2)];   // 32 x 768 bf16 (48 KB)
    __shared__ __align__(16) uint32_t Hd[EPB * (D2 / 2)];   // 32 x 512 bf16 (32 KB)
    __shared__ const float* rA[EPB];   // pos_emb row a0
    __shared__ const float* rB[EPB];   // pos_emb row a1
    __shared__ const float* rP[EPB];   // pred_emb row
    __shared__ const float* rT[EPB];   // "third" row
    __shared__ int   seg[2][EPB];
    __shared__ float wbm[EPB];

    const int tid  = threadIdx.x;
    const int lane = tid & 31;
    const int wv   = tid >> 5;
    const int e0   = blockIdx.x * EPB;

    if (tid < EPB) {
        int e = e0 + tid;
        int b = e / F_;
        int i0 = a0[e], i1 = a1[e], pi = pidx[e], ri = ridx[e];
        bool isr = (pi == 1);
        rA[tid] = pos + i0 * D_;
        rB[tid] = pos + i1 * D_;
        rP[tid] = pemb + pi * D_;
        rT[tid] = isr ? (remb + (ri + 1) * D_) : (pos + i1 * D_);
        seg[0][tid] = b * S_ + (isr ? i0 : i1);   // dst_fwd
        seg[1][tid] = b * S_ + i0;                // dst_bwd
        wbm[tid] = (!isr && pi != 0) ? 1.0f : 0.0f;
    }
    __syncthreads();

    unsigned short* Hs = (unsigned short*)Hd;
    const int lhi = lane >> 4;
    const int lm  = lane & 15;

    for (int p = 0; p < 2; ++p) {
        // ---- stage concatenated input (32 x 768) f32 -> packed bf16 ----
        for (int t = tid; t < EPB * (D3 / 2); t += 256) {
            int m  = t / (D3 / 2);
            int cd = t - m * (D3 / 2);        // dword column
            int part = cd >> 7;               // 128 dwords per 256-f32 part
            int cc = (cd & 127) << 1;
            const float* s;
            if (p == 0) s = (part == 0) ? rA[m] : (part == 1) ? rP[m] : rT[m];
            else        s = (part == 0) ? rB[m] : (part == 1) ? rP[m] : rA[m];
            float2 f = *(const float2*)(s + cc);
            Xd[t] = pack2bf(f.x, f.y);
        }
        __syncthreads();

        // ---- GEMM1: (32x768) @ (768x512): 8 waves x (2 M-tiles x 4 N-tiles)
        {
            v8f acc[8];
#pragma unroll
            for (int i = 0; i < 8; ++i) acc[i] = vzero8();
            wmma_gemm<D3, 4, 2>(Xd, (const uint32_t*)W1t, wv * 64, lane, acc);
#pragma unroll
            for (int mt = 0; mt < 2; ++mt)
#pragma unroll
                for (int j = 0; j < 4; ++j) {
                    int n = wv * 64 + j * 16 + lm;
                    float b1n = b1[n];
#pragma unroll
                    for (int g = 0; g < 8; ++g) {
                        int m = mt * 16 + g + 8 * lhi;
                        float h = gelu_exact(acc[mt * 4 + j][g] + b1n);
                        Hs[m * D2 + n] = f2bf(h);
                    }
                }
        }
        __syncthreads();

        // ---- GEMM2: (32x512) @ (512x256): 8 waves x (2 M x 2 N); scatter
        {
            v8f acc[4];
#pragma unroll
            for (int i = 0; i < 4; ++i) acc[i] = vzero8();
            wmma_gemm<D2, 2, 2>(Hd, (const uint32_t*)W2t, wv * 32, lane, acc);
#pragma unroll
            for (int mt = 0; mt < 2; ++mt)
#pragma unroll
                for (int j = 0; j < 2; ++j) {
                    int n = wv * 32 + j * 16 + lm;
                    float b2n = b2[n];
#pragma unroll
                    for (int g = 0; g < 8; ++g) {
                        int m = mt * 16 + g + 8 * lhi;
                        float w = (p == 0) ? 1.0f : wbm[m];
                        if (w != 0.0f) {
                            float v = (acc[mt * 2 + j][g] + b2n) * w;
                            unsafeAtomicAdd(&agg[seg[p][m] * D_ + n], v);
                        }
                    }
                }
        }
        __syncthreads();
    }
}

// ---------------- layernorm + pool kernel (one (b,s) row per block) ----------------
__global__ __launch_bounds__(256) void norm_pool_kernel(
        const float* __restrict__ pos, const float* __restrict__ agg,
        const float* __restrict__ lng, const float* __restrict__ lnb,
        float* __restrict__ pooled) {
    int r = blockIdx.x;          // 0 .. B*S-1
    int b = r >> 9;
    int s = r & (S_ - 1);
    int d = threadIdx.x;
    float x = pos[s * D_ + d] + agg[(size_t)r * D_ + d];

    float sum = x, sq = x * x;
#pragma unroll
    for (int o = 16; o > 0; o >>= 1) {
        sum += __shfl_xor(sum, o, 32);
        sq  += __shfl_xor(sq,  o, 32);
    }
    __shared__ float ssum[8], ssq[8];
    int wv = threadIdx.x >> 5;
    if ((threadIdx.x & 31) == 0) { ssum[wv] = sum; ssq[wv] = sq; }
    __syncthreads();
    float tot = 0.f, tq = 0.f;
#pragma unroll
    for (int i = 0; i < 8; ++i) { tot += ssum[i]; tq += ssq[i]; }
    float mu  = tot * (1.0f / D_);
    float var = tq * (1.0f / D_) - mu * mu;
    float nv  = (x - mu) * rsqrtf(var + 1e-5f) * lng[d] + lnb[d];
    unsafeAtomicAdd(&pooled[b * D_ + d], nv * (1.0f / S_));
}

// ---------------- head kernel: (16x256)->gelu(512)->512, single block ----------------
__global__ __launch_bounds__(256) void head_kernel(
        const float* __restrict__ pooled,
        const unsigned short* __restrict__ Wl1t, const float* __restrict__ bl1,
        const unsigned short* __restrict__ Wl2t, const float* __restrict__ bl2,
        float* __restrict__ out) {
    __shared__ __align__(16) uint32_t Ad[16 * (D_ / 2)];     // 16x256 bf16
    __shared__ __align__(16) uint32_t Hd[16 * (DLAT / 2)];   // 16x512 bf16
    unsigned short* Hs = (unsigned short*)Hd;
    int tid = threadIdx.x, lane = tid & 31, wv = tid >> 5;
    const int lhi = lane >> 4, lm = lane & 15;

    for (int t = tid; t < 16 * (D_ / 2); t += 256) {
        float2 f = *(const float2*)(pooled + 2 * t);
        Ad[t] = pack2bf(f.x, f.y);
    }
    __syncthreads();

    {   // GEMM1: (16x256) @ (256x512)
        v8f acc[4];
#pragma unroll
        for (int j = 0; j < 4; ++j) acc[j] = vzero8();
        wmma_gemm<D_, 4, 1>(Ad, (const uint32_t*)Wl1t, wv * 64, lane, acc);
#pragma unroll
        for (int j = 0; j < 4; ++j) {
            int n = wv * 64 + j * 16 + lm;
            float bn = bl1[n];
#pragma unroll
            for (int g = 0; g < 8; ++g) {
                int m = g + 8 * lhi;
                Hs[m * DLAT + n] = f2bf(gelu_exact(acc[j][g] + bn));
            }
        }
    }
    __syncthreads();

    {   // GEMM2: (16x512) @ (512x512) -> out
        v8f acc[4];
#pragma unroll
        for (int j = 0; j < 4; ++j) acc[j] = vzero8();
        wmma_gemm<DLAT, 4, 1>(Hd, (const uint32_t*)Wl2t, wv * 64, lane, acc);
#pragma unroll
        for (int j = 0; j < 4; ++j) {
            int n = wv * 64 + j * 16 + lm;
            float bn = bl2[n];
#pragma unroll
            for (int g = 0; g < 8; ++g) {
                int m = g + 8 * lhi;
                out[m * DLAT + n] = acc[j][g] + bn;
            }
        }
    }
}

// ---------------- host-side launch ----------------
extern "C" void kernel_launch(void* const* d_in, const int* in_sizes, int n_in,
                              void* d_out, int out_size, void* d_ws, size_t ws_size,
                              hipStream_t stream) {
    const int*   a0    = (const int*)d_in[0];
    const int*   a1    = (const int*)d_in[1];
    const int*   pidx  = (const int*)d_in[2];
    const int*   ridx  = (const int*)d_in[3];
    /* d_in[4] = seq_len scalar (S_=512, compile-time) */
    const float* pos   = (const float*)d_in[5];
    const float* pemb  = (const float*)d_in[6];
    const float* remb  = (const float*)d_in[7];
    const float* W1    = (const float*)d_in[8];
    const float* b1    = (const float*)d_in[9];
    const float* W2    = (const float*)d_in[10];
    const float* b2    = (const float*)d_in[11];
    const float* lng   = (const float*)d_in[12];
    const float* lnb   = (const float*)d_in[13];
    const float* Wl1   = (const float*)d_in[14];
    const float* bl1   = (const float*)d_in[15];
    const float* Wl2   = (const float*)d_in[16];
    const float* bl2   = (const float*)d_in[17];

    // workspace carve-up
    char* ws = (char*)d_ws;
    float* agg    = (float*)ws;                                   // B*S*D f32 = 8 MB
    float* pooled = (float*)(ws + (size_t)B_ * S_ * D_ * 4);      // B*D f32
    char* wp = ws + (size_t)B_ * S_ * D_ * 4 + (size_t)B_ * D_ * 4;
    unsigned short* W1t  = (unsigned short*)wp;  wp += (size_t)D2 * D3 * 2;    // [512][768]
    unsigned short* W2t  = (unsigned short*)wp;  wp += (size_t)D_ * D2 * 2;    // [256][512]
    unsigned short* Wl1t = (unsigned short*)wp;  wp += (size_t)DLAT * D_ * 2;  // [512][256]
    unsigned short* Wl2t = (unsigned short*)wp;                                 // [512][512]

    // 1) zero accumulators (agg and pooled are contiguous)
    {
        int n = B_ * S_ * D_ + B_ * D_;
        zero_f32_kernel<<<(n + 255) / 256, 256, 0, stream>>>(agg, n);
    }
    // 2) weight convert+transpose to bf16 [N][K] (LDS-tiled)
    transpose_bf16_kernel<<<(D3 / 32) * (D2 / 32), 256, 0, stream>>>(W1, W1t, D3, D2);
    transpose_bf16_kernel<<<(D2 / 32) * (D_ / 32), 256, 0, stream>>>(W2, W2t, D2, D_);
    transpose_bf16_kernel<<<(D_ / 32) * (DLAT / 32), 256, 0, stream>>>(Wl1, Wl1t, D_, DLAT);
    transpose_bf16_kernel<<<(DLAT / 32) * (DLAT / 32), 256, 0, stream>>>(Wl2, Wl2t, DLAT, DLAT);
    // 3) edge MLPs (fwd+bwd) with WMMA + atomic scatter
    edge_mlp_kernel<<<(B_ * F_) / EPB, 256, 0, stream>>>(
        a0, a1, pidx, ridx, pos, pemb, remb, W1t, b1, W2t, b2, agg);
    // 4) layernorm + mean pool
    norm_pool_kernel<<<B_ * S_, 256, 0, stream>>>(pos, agg, lng, lnb, pooled);
    // 5) latent head
    head_kernel<<<1, 256, 0, stream>>>(pooled, Wl1t, bl1, Wl2t, bl2, (float*)d_out);
    (void)in_sizes; (void)n_in; (void)out_size; (void)ws_size;
}
